// GNNEncoder_103079215831
// MI455X (gfx1250) — matrix operand order repaired and checked by reading, not compiled, not run on hardware
//
#include <hip/hip_runtime.h>
#include <hip/hip_bf16.h>

// ---------------------------------------------------------------------------
// GAT encoder for MI455X (gfx1250, wave32).
// GEMMs via v_wmma_f32_16x16x32_bf16 (fp32 -> bf16 staged in LDS, fp32 accum).
// Segment softmax / aggregation via fp32 global atomics (L2-resident).
// ---------------------------------------------------------------------------

typedef __attribute__((ext_vector_type(16))) __bf16          v16bf;
typedef __attribute__((ext_vector_type(8)))  float           v8f;
typedef __attribute__((ext_vector_type(8)))  unsigned short  us8;
typedef __attribute__((ext_vector_type(2)))  unsigned int    ui2;

union Frag16 { v16bf v; us8 h[2]; unsigned short u[16]; };

__device__ __forceinline__ unsigned short f2bf(float f) {
  unsigned int u = __float_as_uint(f);
  u += 0x7FFFu + ((u >> 16) & 1u);          // round-to-nearest-even
  return (unsigned short)(u >> 16);
}
__device__ __forceinline__ unsigned int pack2bf(float a, float b) {
  return (unsigned int)f2bf(a) | ((unsigned int)f2bf(b) << 16);
}

// monotonic float <-> uint key for atomicMax-based segment max
__device__ __forceinline__ unsigned int fkey(float f) {
  unsigned int u = __float_as_uint(f);
  return (u & 0x80000000u) ? ~u : (u | 0x80000000u);
}
__device__ __forceinline__ float funkey(unsigned int k) {
  unsigned int u = (k & 0x80000000u) ? (k ^ 0x80000000u) : ~k;
  return __uint_as_float(u);
}
__device__ __forceinline__ float lrelu(float x) { return x > 0.f ? x : 0.2f * x; }

// ---------------------------------------------------------------------------
// C[M,N] = A[M,K] * W[K,N]; N,K compile-time so the WMMA tile loop fully
// unrolls (static accumulator VGPRs, no movrel) and LDS fragment reads are
// contiguous b128 loads.
// Block: 256 threads = 8 waves; block tile 32 rows x N.
// Waves 0-3 -> rows 0..15, waves 4-7 -> rows 16..31; wave owns N/4 columns.
// ---------------------------------------------------------------------------
template <int N, int K>
__global__ __launch_bounds__(256) void gemm_wmma_bf16(
    const float* __restrict__ A, const float* __restrict__ W,
    float* __restrict__ C, int M)
{
  __shared__ unsigned short lA[32][32];     // 32 rows x 32 K (bf16 bits)
  __shared__ unsigned short lBt[N][32];     // B transposed: [col][k]

  constexpr int NT = N / 64;                // WMMA tiles per wave (4 or 2)
  constexpr int LOGN = (N == 256) ? 8 : 7;

  const int tid      = threadIdx.x;
  const int lane     = tid & 31;
  const int wave     = tid >> 5;            // 0..7
  const int waveRow  = (wave >> 2) << 4;    // 0 or 16
  const int colBase  = (wave & 3) * (N >> 2);
  const int rowBlock = blockIdx.x << 5;

  // A staging map: each thread owns one float4 of the 32x32 tile
  const int ar  = tid >> 3;                 // 0..31
  const int ac4 = (tid & 7) << 2;           // 0,4,...,28
  const int agr = rowBlock + ar;

  v8f acc[NT] = {};

  for (int kk = 0; kk < K; kk += 32) {
    // ---- stage A tile (32x32) : one float4 load + one uint2 LDS store ----
    {
      float4 f = make_float4(0.f, 0.f, 0.f, 0.f);
      if (agr < M) f = *(const float4*)&A[(long)agr * K + kk + ac4];
      ui2 p; p.x = pack2bf(f.x, f.y); p.y = pack2bf(f.z, f.w);
      *(ui2*)&lA[ar][ac4] = p;
    }
    // ---- stage B tile (32 x N) transposed: unit = (col, 4 consecutive k) ----
#pragma unroll
    for (int u0 = tid; u0 < N * 8; u0 += 256) {
      int col = u0 & (N - 1);
      int k0  = (u0 >> LOGN) << 2;
      float w0 = W[(long)(kk + k0 + 0) * N + col];
      float w1 = W[(long)(kk + k0 + 1) * N + col];
      float w2 = W[(long)(kk + k0 + 2) * N + col];
      float w3 = W[(long)(kk + k0 + 3) * N + col];
      ui2 p; p.x = pack2bf(w0, w1); p.y = pack2bf(w2, w3);
      *(ui2*)&lBt[col][k0] = p;
    }
    if (kk + 32 < K && agr < M)
      __builtin_prefetch(&A[(long)agr * K + kk + 32 + ac4], 0, 1);
    __syncthreads();

    // ---- A fragment: lanes 0-15 K{kb..kb+7, 16+kb..}, kb=0; lanes 16-31 kb=8
    Frag16 af;
    {
      int r  = waveRow + (lane & 15);
      int kb = (lane >> 4) << 3;            // 0 or 8
      af.h[0] = *(const us8*)&lA[r][kb];
      af.h[1] = *(const us8*)&lA[r][16 + kb];
    }
    // ---- B fragments (two 16B runs per lane: k kb..kb+7, kb+16..kb+23) + WMMA, fully unrolled ----
#pragma unroll
    for (int t = 0; t < NT; ++t) {
      Frag16 bfm;
      int col = colBase + (t << 4) + (lane & 15);
      int kb  = (lane >> 4) << 3;           // 0 or 8
      bfm.h[0] = *(const us8*)&lBt[col][kb];
      bfm.h[1] = *(const us8*)&lBt[col][kb + 16];
      acc[t] = __builtin_amdgcn_wmma_f32_16x16x32_bf16(
          false, af.v, false, bfm.v, (short)0, acc[t], false, false);
    }
    __syncthreads();
  }

  // ---- writeback: VGPR g -> M = g + 8*(lane>=16), N = lane&15 ----
#pragma unroll
  for (int t = 0; t < NT; ++t) {
    int colT = colBase + (t << 4) + (lane & 15);
#pragma unroll
    for (int g = 0; g < 8; ++g) {
      int gr = rowBlock + waveRow + g + ((lane >> 4) << 3);
      if (gr < M) C[(long)gr * N + colT] = acc[t][g];
    }
  }
}

// ---------------------------------------------------------------------------
__global__ void zero_f32(float* __restrict__ p, long n) {
  long i = (long)blockIdx.x * blockDim.x + threadIdx.x;
  long st = (long)gridDim.x * blockDim.x;
  for (; i < n; i += st) p[i] = 0.f;
}

// per-(node,head) attention scores: al_s = h . a_src, al_d = h . a_dst
__global__ void node_scores(const float* __restrict__ h,
                            const float* __restrict__ a_src,
                            const float* __restrict__ a_dst,
                            float* __restrict__ al_s, float* __restrict__ al_d,
                            int n, int H, int C) {
  int i = blockIdx.x * blockDim.x + threadIdx.x;
  if (i >= n * H) return;
  int hh = i % H;
  const float* hp = h + (long)i * C;
  const float* as = a_src + hh * C;
  const float* ad = a_dst + hh * C;
  float s = 0.f, d = 0.f;
  for (int c = 0; c < C; ++c) { float v = hp[c]; s += v * as[c]; d += v * ad[c]; }
  al_s[i] = s; al_d[i] = d;
}

// segment max over dst (atomicMax on monotonic uint keys); self-loops appended
__global__ void edge_max(const int* __restrict__ src, const int* __restrict__ dst,
                         int E, int n,
                         const float* __restrict__ al_s, const float* __restrict__ al_d,
                         unsigned int* __restrict__ mkey, int H) {
  int i = blockIdx.x * blockDim.x + threadIdx.x;
  if (i >= E + n) return;
  int s, d;
  if (i < E) { s = src[i]; d = dst[i]; } else { s = d = i - E; }
  for (int hh = 0; hh < H; ++hh) {
    float e = lrelu(al_s[s * H + hh] + al_d[d * H + hh]);
    atomicMax(&mkey[d * H + hh], fkey(e));
  }
}

// segment sum of exp(e - m)
__global__ void edge_den(const int* __restrict__ src, const int* __restrict__ dst,
                         int E, int n,
                         const float* __restrict__ al_s, const float* __restrict__ al_d,
                         const unsigned int* __restrict__ mkey,
                         float* __restrict__ den, int H) {
  int i = blockIdx.x * blockDim.x + threadIdx.x;
  if (i >= E + n) return;
  int s, d;
  if (i < E) { s = src[i]; d = dst[i]; } else { s = d = i - E; }
  for (int hh = 0; hh < H; ++hh) {
    float e = lrelu(al_s[s * H + hh] + al_d[d * H + hh]);
    float m = funkey(mkey[d * H + hh]);
    atomicAdd(&den[d * H + hh], __expf(e - m));
  }
}

// one wave per edge: out[dst] += alpha * h[src]  (lanes parallel over channels)
__global__ __launch_bounds__(256) void edge_agg(
    const int* __restrict__ src, const int* __restrict__ dst, int E, int n,
    const float* __restrict__ al_s, const float* __restrict__ al_d,
    const unsigned int* __restrict__ mkey, const float* __restrict__ den,
    const float* __restrict__ h, float* __restrict__ out, int H, int C) {
  int lane = threadIdx.x & 31;
  long gw = (long)blockIdx.x * (blockDim.x >> 5) + (threadIdx.x >> 5);
  long nw = (long)gridDim.x * (blockDim.x >> 5);
  for (long i = gw; i < (long)E + n; i += nw) {
    int s, d;
    if (i < E) { s = src[i]; d = dst[i]; } else { s = d = (int)(i - E); }
    for (int hh = 0; hh < H; ++hh) {
      float e = lrelu(al_s[s * H + hh] + al_d[d * H + hh]);
      float m = funkey(mkey[d * H + hh]);
      float alpha = __expf(e - m) / den[d * H + hh];
      const float* hs = h + ((long)s * H + hh) * C;
      float* od = out + ((long)d * H + hh) * C;
      for (int c = lane; c < C; c += 32)
        atomicAdd(&od[c], alpha * hs[c]);
    }
  }
}

// x = relu(x + bias)  (layer-1 epilogue, in place)
__global__ void bias_relu(float* __restrict__ x, const float* __restrict__ bias,
                          int n, int N) {
  long i = (long)blockIdx.x * blockDim.x + threadIdx.x;
  if (i >= (long)n * N) return;
  float v = x[i] + bias[i % N];
  x[i] = v > 0.f ? v : 0.f;
}

__global__ void pool_cnt(const int* __restrict__ batch, float* __restrict__ cnt, int n) {
  int i = blockIdx.x * blockDim.x + threadIdx.x;
  if (i < n) atomicAdd(&cnt[batch[i]], 1.f);
}

__global__ void pool_accum(const float* __restrict__ h, const int* __restrict__ batch,
                           float* __restrict__ sums, int n, int N) {
  long i = (long)blockIdx.x * blockDim.x + threadIdx.x;
  if (i >= (long)n * N) return;
  int node = (int)(i / N), c = (int)(i - (long)node * N);
  atomicAdd(&sums[batch[node] * N + c], h[i]);
}

__global__ void pool_final(const float* __restrict__ sums, const float* __restrict__ cnt,
                           const float* __restrict__ b, float* __restrict__ out,
                           int G, int N) {
  int i = blockIdx.x * blockDim.x + threadIdx.x;
  if (i >= G * N) return;
  int g = i / N, c = i - g * N;
  float k = cnt[g];
  out[i] = (k > 0.5f) ? (sums[i] / k + b[c]) : 0.f;
}

// ---------------------------------------------------------------------------
extern "C" void kernel_launch(void* const* d_in, const int* in_sizes, int n_in,
                              void* d_out, int out_size, void* d_ws, size_t ws_size,
                              hipStream_t stream) {
  const float* x      = (const float*)d_in[0];
  const int*   src    = (const int*)  d_in[1];
  const int*   dst    = (const int*)  d_in[2];
  const int*   batch  = (const int*)  d_in[3];
  const float* W1     = (const float*)d_in[4];
  const float* a_src1 = (const float*)d_in[5];
  const float* a_dst1 = (const float*)d_in[6];
  const float* b1     = (const float*)d_in[7];
  const float* W2     = (const float*)d_in[8];
  const float* a_src2 = (const float*)d_in[9];
  const float* a_dst2 = (const float*)d_in[10];
  const float* b2     = (const float*)d_in[11];

  const int n  = in_sizes[3];          // 50000 nodes
  const int E  = in_sizes[1];          // 800000 edges (self-loops appended on GPU)
  const int EE = E + n;
  const int D1 = 256, H1 = 4, C1 = 64, D2 = 128, G = 64;

  float* ws = (float*)d_ws;
  size_t offH   = 0;                                   // h1 [n,256] (reused as h2 [n,128])
  size_t offO   = offH   + (size_t)n * D1;             // out1 accumulator / hrelu [n,256]
  size_t offAS  = offO   + (size_t)n * D1;             // al_s [n,4]
  size_t offAD  = offAS  + (size_t)n * H1;             // al_d [n,4]
  size_t offM   = offAD  + (size_t)n * H1;             // max keys (uint) [n,4]
  size_t offDen = offM   + (size_t)n * H1;             // softmax denom [n,4]
  size_t offO2  = offDen + (size_t)n * H1;             // out2 [n,128]
  size_t offS   = offO2  + (size_t)n * D2;             // pool sums [64,128]
  size_t offC   = offS   + (size_t)G * D2;             // pool counts [64]
  size_t total  = offC + G;
  (void)ws_size; (void)n_in; (void)out_size;

  const int T = 256;
  // 1) zero all accumulator / softmax / pooling state
  zero_f32<<<8192, T, 0, stream>>>(ws + offO, (long)(total - offO));

  // ---------------- layer 1 (H=4, C=64): h1 = x @ W1 ----------------
  gemm_wmma_bf16<256, 128><<<(n + 31) / 32, T, 0, stream>>>(x, W1, ws + offH, n);
  node_scores<<<(n * H1 + T - 1) / T, T, 0, stream>>>(ws + offH, a_src1, a_dst1,
                                                      ws + offAS, ws + offAD, n, H1, C1);
  edge_max<<<(EE + T - 1) / T, T, 0, stream>>>(src, dst, E, n, ws + offAS, ws + offAD,
                                               (unsigned int*)(ws + offM), H1);
  edge_den<<<(EE + T - 1) / T, T, 0, stream>>>(src, dst, E, n, ws + offAS, ws + offAD,
                                               (const unsigned int*)(ws + offM),
                                               ws + offDen, H1);
  edge_agg<<<(EE + 7) / 8, T, 0, stream>>>(src, dst, E, n, ws + offAS, ws + offAD,
                                           (const unsigned int*)(ws + offM), ws + offDen,
                                           ws + offH, ws + offO, H1, C1);
  bias_relu<<<((long)n * D1 + T - 1) / T, T, 0, stream>>>(ws + offO, b1, n, D1);

  // re-zero softmax state for layer 2 (mkey + den are contiguous)
  zero_f32<<<512, T, 0, stream>>>(ws + offM, (long)(2 * (size_t)n * H1));

  // ---------------- layer 2 (H=1, C=128): h2 = hrelu @ W2 ----------------
  gemm_wmma_bf16<128, 256><<<(n + 31) / 32, T, 0, stream>>>(ws + offO, W2, ws + offH, n);
  node_scores<<<(n + T - 1) / T, T, 0, stream>>>(ws + offH, a_src2, a_dst2,
                                                 ws + offAS, ws + offAD, n, 1, D2);
  edge_max<<<(EE + T - 1) / T, T, 0, stream>>>(src, dst, E, n, ws + offAS, ws + offAD,
                                               (unsigned int*)(ws + offM), 1);
  edge_den<<<(EE + T - 1) / T, T, 0, stream>>>(src, dst, E, n, ws + offAS, ws + offAD,
                                               (const unsigned int*)(ws + offM),
                                               ws + offDen, 1);
  edge_agg<<<(EE + 7) / 8, T, 0, stream>>>(src, dst, E, n, ws + offAS, ws + offAD,
                                           (const unsigned int*)(ws + offM), ws + offDen,
                                           ws + offH, ws + offO2, 1, D2);

  // ---------------- global mean pool ----------------
  pool_cnt<<<(n + T - 1) / T, T, 0, stream>>>(batch, ws + offC, n);
  pool_accum<<<((long)n * D2 + T - 1) / T, T, 0, stream>>>(ws + offO2, batch, ws + offS, n, D2);
  pool_final<<<(G * D2 + T - 1) / T, T, 0, stream>>>(ws + offS, ws + offC, b2,
                                                     (float*)d_out, G, D2);
}
